// DeepseekV3MoE_88630945120717
// MI455X (gfx1250) — compile-verified, matrix operand order checked by reading, not run on hardware
//
#include <hip/hip_runtime.h>
#include <hip/hip_bf16.h>

// ---------------------------------------------------------------------------
// Problem constants (from reference)
// ---------------------------------------------------------------------------
constexpr int Bc   = 2;
constexpr int Sc   = 2048;
constexpr int Hc   = 2048;     // hidden
constexpr int Ec   = 32;       // routed experts
constexpr int Gc   = 8;        // groups
constexpr int TKGc = 4;        // topk groups
constexpr int Kc   = 8;        // experts per token
constexpr int Ic   = 1024;     // moe intermediate
constexpr int SIc  = 2048;     // shared intermediate
constexpr int Tc   = Bc * Sc;  // 4096 tokens
constexpr int TKc  = Tc * Kc;  // 32768 dispatched rows
constexpr int MAXT = 576;      // >= max number of 64-row expert tiles (512+32)
constexpr float ROUTED_SCALE = 2.5f;

// ---------------------------------------------------------------------------
// bf16 helpers (bit-level, RNE) + WMMA vector types
// ---------------------------------------------------------------------------
typedef __bf16 bf16x16 __attribute__((ext_vector_type(16)));
typedef float  f32x8   __attribute__((ext_vector_type(8)));

union Frag { bf16x16 v; uint4 q[2]; };

__device__ __forceinline__ unsigned short f2bf(float f) {
  unsigned u = __float_as_uint(f);
  u += 0x7fffu + ((u >> 16) & 1u);
  return (unsigned short)(u >> 16);
}
__device__ __forceinline__ float bf2f(unsigned short b) {
  return __uint_as_float(((unsigned)b) << 16);
}
__device__ __forceinline__ unsigned pack2bf(float lo, float hi) {
  return (unsigned)f2bf(lo) | ((unsigned)f2bf(hi) << 16);
}
__device__ __forceinline__ int imin(int a, int b) { return a < b ? a : b; }
// fast sigmoid / silu: single v_rcp_f32 (~1 ulp), well below bf16 rounding
__device__ __forceinline__ float fast_sigmoid(float x) {
  return __builtin_amdgcn_rcpf(1.f + __expf(-x));
}
__device__ __forceinline__ float silu(float x) { return x * fast_sigmoid(x); }
__device__ __forceinline__ float f4get(const float4& v, int j) {
  return j == 0 ? v.x : j == 1 ? v.y : j == 2 ? v.z : v.w;
}

#define WMMA_BF16(A, B, C) \
  __builtin_amdgcn_wmma_f32_16x16x32_bf16(false, (A), false, (B), (short)0, (C), false, false)

// ---------------------------------------------------------------------------
// 1) Router: one wave32 per token, lane == expert
// ---------------------------------------------------------------------------
__global__ __launch_bounds__(256) void router_kernel(
    const float* __restrict__ x, const float* __restrict__ rw,
    const float* __restrict__ bias, int* __restrict__ topk_idx,
    float* __restrict__ topk_w) {
  const int wl    = threadIdx.x >> 5;                    // wave in block (0..7)
  const int lane  = threadIdx.x & 31;                    // expert id
  const int token = blockIdx.x * 8 + wl;

  const float* xt = x + (size_t)token * Hc;
  float acc = 0.f;
  for (int h = 0; h < Hc; h += 4) {
    float4 xv = *(const float4*)(xt + h);
    acc += xv.x * rw[(h + 0) * Ec + lane];
    acc += xv.y * rw[(h + 1) * Ec + lane];
    acc += xv.z * rw[(h + 2) * Ec + lane];
    acc += xv.w * rw[(h + 3) * Ec + lane];
  }
  float score = fast_sigmoid(acc);
  float sb    = score + bias[lane];

  __shared__ float s_sc[8][32];
  __shared__ float s_sb[8][32];
  s_sc[wl][lane] = score;
  s_sb[wl][lane] = sb;
  __syncthreads();

  if (lane == 0) {
    // group scores: sum of top-2 inside each group of 4 consecutive experts
    float gsc[Gc];
    for (int g = 0; g < Gc; ++g) {
      float m1 = -1e30f, m2 = -1e30f;
      for (int j = 0; j < Ec / Gc; ++j) {
        float v = s_sb[wl][g * (Ec / Gc) + j];
        if (v > m1) { m2 = m1; m1 = v; }
        else if (v > m2) { m2 = v; }
      }
      gsc[g] = m1 + m2;
    }
    // pick top TKG groups (ties -> lower index, matching jax top_k)
    bool gsel[Gc] = {};
    for (int it = 0; it < TKGc; ++it) {
      int bg = -1; float bv = -1e30f;
      for (int g = 0; g < Gc; ++g)
        if (!gsel[g] && gsc[g] > bv) { bv = gsc[g]; bg = g; }
      gsel[bg] = true;
    }
    // mask scores, pick top K experts
    float masked[Ec];
    for (int e = 0; e < Ec; ++e)
      masked[e] = gsel[e / (Ec / Gc)] ? s_sb[wl][e] : 0.0f;
    bool used[Ec] = {};
    int   idxs[Kc];
    float wv[Kc];
    float wsum = 0.f;
    for (int it = 0; it < Kc; ++it) {
      int be = -1; float bv = -1e30f;
      for (int e = 0; e < Ec; ++e)
        if (!used[e] && masked[e] > bv) { bv = masked[e]; be = e; }
      used[be] = true;
      idxs[it] = be;
      wv[it]   = s_sc[wl][be];   // take_along on raw sigmoid scores
      wsum += wv[it];
    }
    float inv = ROUTED_SCALE / wsum;
    for (int k = 0; k < Kc; ++k) {
      topk_idx[token * Kc + k] = idxs[k];
      topk_w[token * Kc + k]   = wv[k] * inv;
    }
  }
}

// ---------------------------------------------------------------------------
// 2) Dispatch bookkeeping
// ---------------------------------------------------------------------------
__global__ void init_kernel(int* counts) {
  if (threadIdx.x < Ec) counts[threadIdx.x] = 0;
}

__global__ void count_kernel(const int* __restrict__ topk_idx, int* counts) {
  int i = blockIdx.x * blockDim.x + threadIdx.x;
  if (i < TKc) atomicAdd(&counts[topk_idx[i]], 1);
}

__global__ void scan_build_kernel(const int* __restrict__ counts, int* offsets,
                                  int* fill, int* tile_meta, int* num_tiles) {
  if (threadIdx.x != 0) return;
  int off = 0, nt = 0;
  for (int e = 0; e < Ec; ++e) {
    offsets[e] = off;
    fill[e]    = off;
    int c = counts[e];
    for (int m = 0; m < c; m += 64) {
      tile_meta[nt * 3 + 0] = e;
      tile_meta[nt * 3 + 1] = off + m;
      tile_meta[nt * 3 + 2] = imin(64, c - m);
      ++nt;
    }
    off += c;
  }
  offsets[Ec] = off;
  *num_tiles  = nt;
}

__global__ void scatter_kernel(const int* __restrict__ topk_idx,
                               const float* __restrict__ topk_w, int* fill,
                               int* row_token, float* pos_weight, int* pos_of) {
  int i = blockIdx.x * blockDim.x + threadIdx.x;
  if (i >= TKc) return;
  int e   = topk_idx[i];
  int pos = atomicAdd(&fill[e], 1);
  row_token[pos]  = i / Kc;
  pos_weight[pos] = topk_w[i];
  pos_of[i]       = pos;
}

// ---------------------------------------------------------------------------
// 3) Fused gate+up GEMM with SwiGLU epilogue (bf16 WMMA, fp32 accumulate).
//    Tail rows are handled by clamping the source row (garbage only reaches
//    discarded C rows) so the k-loop has no divergent branches. Global loads
//    for the next k-step are issued before the first barrier (sw pipeline).
// ---------------------------------------------------------------------------
template <bool GATHER>
__global__ __launch_bounds__(256) void gateup_kernel(
    const float* __restrict__ x, const float* __restrict__ Wg_all,
    const float* __restrict__ Wu_all, const int* __restrict__ tile_meta,
    const int* __restrict__ num_tiles, const int* __restrict__ row_token,
    unsigned short* __restrict__ Hout, int Kdim, int Ncols, int Mtotal) {
  int e, pos_base, rows;
  if (GATHER) {
    if ((int)blockIdx.x >= *num_tiles) return;
    e        = tile_meta[blockIdx.x * 3 + 0];
    pos_base = tile_meta[blockIdx.x * 3 + 1];
    rows     = tile_meta[blockIdx.x * 3 + 2];
  } else {
    e        = 0;
    pos_base = blockIdx.x * 64;
    rows     = imin(64, Mtotal - pos_base);
  }
  const float* Wg = Wg_all + (size_t)e * Kdim * Ncols;
  const float* Wu = Wu_all + (size_t)e * Kdim * Ncols;
  const int nb = blockIdx.y * 64;

  __shared__ __align__(16) unsigned short sA[64 * 40];
  __shared__ __align__(16) unsigned short sBg[64 * 40];
  __shared__ __align__(16) unsigned short sBu[64 * 40];

  const int tid  = threadIdx.x;
  const int lane = tid & 31, w = tid >> 5;
  const int wm = w >> 2, wn = w & 3;         // 2x4 wave grid
  const int lr = lane & 15, hi = lane >> 4;
  const int kb = hi * 8;                      // K-half select for fragments

  f32x8 cg[2] = {};
  f32x8 cu[2] = {};

  // A staging: thread -> (row, 8 consecutive k); row clamped for tails
  const int ar  = tid >> 2;
  const int ak  = (tid & 3) * 8;
  const int arc = imin(ar, rows - 1);
  int arow;
  if (GATHER) arow = row_token[pos_base + arc];
  else        arow = pos_base + arc;
  const float* Aptr = x + (size_t)arow * Kdim + ak;

  // B staging: thread -> (2 consecutive k rows, 4 consecutive n cols);
  // packed b32 LDS stores into transposed [n][k] layout.
  const int nn    = (tid & 15) * 4;   // 0..60
  const int kpair = (tid >> 4) * 2;   // 0..30

  // prologue: global loads for k0 = 0
  float4 af0 = *(const float4*)(Aptr + 0);
  float4 af1 = *(const float4*)(Aptr + 4);
  const float* gp = Wg + (size_t)kpair * Ncols + nb + nn;
  const float* up = Wu + (size_t)kpair * Ncols + nb + nn;
  float4 gg0 = *(const float4*)gp, gg1 = *(const float4*)(gp + Ncols);
  float4 uu0 = *(const float4*)up, uu1 = *(const float4*)(up + Ncols);

  for (int k0 = 0; k0 < Kdim; k0 += 32) {
    // --- stage current tile to LDS ---
    uint4 pk;
    pk.x = pack2bf(af0.x, af0.y); pk.y = pack2bf(af0.z, af0.w);
    pk.z = pack2bf(af1.x, af1.y); pk.w = pack2bf(af1.z, af1.w);
    *(uint4*)&sA[ar * 40 + ak] = pk;
#pragma unroll
    for (int j = 0; j < 4; ++j) {
      *(unsigned*)&sBg[(nn + j) * 40 + kpair] =
          pack2bf(f4get(gg0, j), f4get(gg1, j));
      *(unsigned*)&sBu[(nn + j) * 40 + kpair] =
          pack2bf(f4get(uu0, j), f4get(uu1, j));
    }

    // --- prefetch next tile (clamped, branch-free) ---
    const int kn = (k0 + 32 < Kdim) ? k0 + 32 : 0;
    af0 = *(const float4*)(Aptr + kn);
    af1 = *(const float4*)(Aptr + kn + 4);
    const float* gpn = Wg + (size_t)(kn + kpair) * Ncols + nb + nn;
    const float* upn = Wu + (size_t)(kn + kpair) * Ncols + nb + nn;
    gg0 = *(const float4*)gpn; gg1 = *(const float4*)(gpn + Ncols);
    uu0 = *(const float4*)upn; uu1 = *(const float4*)(upn + Ncols);

    __syncthreads();

    // --- fragments + WMMA ---
    Frag bg, bu, a0, a1;
    const unsigned short* pb = &sBg[(wn * 16 + lr) * 40 + kb];
    bg.q[0] = *(const uint4*)pb;  bg.q[1] = *(const uint4*)(pb + 16);
    pb = &sBu[(wn * 16 + lr) * 40 + kb];
    bu.q[0] = *(const uint4*)pb;  bu.q[1] = *(const uint4*)(pb + 16);
    const unsigned short* pa = &sA[(wm * 32 + lr) * 40 + kb];
    a0.q[0] = *(const uint4*)pa;  a0.q[1] = *(const uint4*)(pa + 16);
    pa = &sA[(wm * 32 + 16 + lr) * 40 + kb];
    a1.q[0] = *(const uint4*)pa;  a1.q[1] = *(const uint4*)(pa + 16);

    cg[0] = WMMA_BF16(a0.v, bg.v, cg[0]);
    cu[0] = WMMA_BF16(a0.v, bu.v, cu[0]);
    cg[1] = WMMA_BF16(a1.v, bg.v, cg[1]);
    cu[1] = WMMA_BF16(a1.v, bu.v, cu[1]);
    __syncthreads();
  }

  // SwiGLU epilogue -> bf16 H buffer
  const int ncol = nb + wn * 16 + lr;
#pragma unroll
  for (int s = 0; s < 2; ++s) {
#pragma unroll
    for (int v = 0; v < 8; ++v) {
      int m = wm * 32 + s * 16 + v + 8 * hi;
      if (m < rows) {
        float h = silu(cg[s][v]) * cu[s][v];
        Hout[(size_t)(pos_base + m) * Ncols + ncol] = f2bf(h);
      }
    }
  }
}

// ---------------------------------------------------------------------------
// 4) Down-projection GEMM. A is already bf16 (H buffers).
//    ROUTED=true : grouped, epilogue scales by per-row routing weight -> bf16
//    ROUTED=false: shared expert, epilogue plain fp32 store to out
// ---------------------------------------------------------------------------
template <bool ROUTED>
__global__ __launch_bounds__(256) void down_kernel(
    const unsigned short* __restrict__ Hin, const float* __restrict__ Wd_all,
    const int* __restrict__ tile_meta, const int* __restrict__ num_tiles,
    const float* __restrict__ pos_weight, unsigned short* __restrict__ OutBf,
    float* __restrict__ OutF, int Kdim, int Ncols, int Mtotal) {
  int e, pos_base, rows;
  if (ROUTED) {
    if ((int)blockIdx.x >= *num_tiles) return;
    e        = tile_meta[blockIdx.x * 3 + 0];
    pos_base = tile_meta[blockIdx.x * 3 + 1];
    rows     = tile_meta[blockIdx.x * 3 + 2];
  } else {
    e        = 0;
    pos_base = blockIdx.x * 64;
    rows     = imin(64, Mtotal - pos_base);
  }
  const float* Wd = Wd_all + (ROUTED ? (size_t)e * Kdim * Ncols : 0);
  const int nb = blockIdx.y * 64;

  __shared__ __align__(16) unsigned short sA[64 * 40];
  __shared__ __align__(16) unsigned short sB[64 * 40];

  const int tid  = threadIdx.x;
  const int lane = tid & 31, w = tid >> 5;
  const int wm = w >> 2, wn = w & 3;
  const int lr = lane & 15, hi = lane >> 4;
  const int kb = hi * 8;

  f32x8 c[2] = {};

  const int ar  = tid >> 2;
  const int ak  = (tid & 3) * 8;
  const int arc = imin(ar, rows - 1);
  const unsigned short* Aptr = Hin + (size_t)(pos_base + arc) * Kdim + ak;

  const int nn    = (tid & 15) * 4;
  const int kpair = (tid >> 4) * 2;

  uint4 av = *(const uint4*)(Aptr + 0);
  const float* dp = Wd + (size_t)kpair * Ncols + nb + nn;
  float4 dd0 = *(const float4*)dp, dd1 = *(const float4*)(dp + Ncols);

  for (int k0 = 0; k0 < Kdim; k0 += 32) {
    *(uint4*)&sA[ar * 40 + ak] = av;
#pragma unroll
    for (int j = 0; j < 4; ++j) {
      *(unsigned*)&sB[(nn + j) * 40 + kpair] =
          pack2bf(f4get(dd0, j), f4get(dd1, j));
    }

    const int kn = (k0 + 32 < Kdim) ? k0 + 32 : 0;
    av = *(const uint4*)(Aptr + kn);
    const float* dpn = Wd + (size_t)(kn + kpair) * Ncols + nb + nn;
    dd0 = *(const float4*)dpn; dd1 = *(const float4*)(dpn + Ncols);

    __syncthreads();

    Frag b, a0, a1;
    const unsigned short* pb = &sB[(wn * 16 + lr) * 40 + kb];
    b.q[0] = *(const uint4*)pb;  b.q[1] = *(const uint4*)(pb + 16);
    const unsigned short* pa = &sA[(wm * 32 + lr) * 40 + kb];
    a0.q[0] = *(const uint4*)pa; a0.q[1] = *(const uint4*)(pa + 16);
    pa = &sA[(wm * 32 + 16 + lr) * 40 + kb];
    a1.q[0] = *(const uint4*)pa; a1.q[1] = *(const uint4*)(pa + 16);

    c[0] = WMMA_BF16(a0.v, b.v, c[0]);
    c[1] = WMMA_BF16(a1.v, b.v, c[1]);
    __syncthreads();
  }

  const int ncol = nb + wn * 16 + lr;
#pragma unroll
  for (int s = 0; s < 2; ++s) {
#pragma unroll
    for (int v = 0; v < 8; ++v) {
      int m = wm * 32 + s * 16 + v + 8 * hi;
      if (m < rows) {
        float val = c[s][v];
        if (ROUTED) {
          float wgt = pos_weight[pos_base + m];
          OutBf[(size_t)(pos_base + m) * Ncols + ncol] = f2bf(val * wgt);
        } else {
          OutF[(size_t)(pos_base + m) * Ncols + ncol] = val;
        }
      }
    }
  }
}

// ---------------------------------------------------------------------------
// 5) Combine: out (already = shared expert result) += sum_k routed[k]
//    fixed k-order accumulation -> deterministic
// ---------------------------------------------------------------------------
__global__ void combine_kernel(const int* __restrict__ pos_of,
                               const unsigned short* __restrict__ down_buf,
                               float* __restrict__ out) {
  int i = blockIdx.x * blockDim.x + threadIdx.x;
  if (i >= Tc * Hc) return;
  int t = i / Hc, h = i % Hc;
  float acc = out[i];
  const int* p = pos_of + t * Kc;
#pragma unroll
  for (int k = 0; k < Kc; ++k)
    acc += bf2f(down_buf[(size_t)p[k] * Hc + h]);
  out[i] = acc;
}

// ---------------------------------------------------------------------------
// launch
// ---------------------------------------------------------------------------
extern "C" void kernel_launch(void* const* d_in, const int* in_sizes, int n_in,
                              void* d_out, int out_size, void* d_ws,
                              size_t ws_size, hipStream_t stream) {
  (void)in_sizes; (void)n_in; (void)out_size; (void)ws_size;
  const float* x           = (const float*)d_in[0];
  const float* router_w    = (const float*)d_in[1];
  const float* router_bias = (const float*)d_in[2];
  const float* w_gate      = (const float*)d_in[3];
  const float* w_up        = (const float*)d_in[4];
  const float* w_down      = (const float*)d_in[5];
  const float* sw_gate     = (const float*)d_in[6];
  const float* sw_up       = (const float*)d_in[7];
  const float* sw_down     = (const float*)d_in[8];
  float* out = (float*)d_out;

  char* p = (char*)d_ws;
  auto carve = [&](size_t bytes) -> char* {
    char* r = p;
    p += (bytes + 255) & ~(size_t)255;
    return r;
  };
  int*            topk_idx   = (int*)carve((size_t)TKc * 4);
  float*          topk_w     = (float*)carve((size_t)TKc * 4);
  int*            counts     = (int*)carve(Ec * 4);
  int*            offsets    = (int*)carve((Ec + 1) * 4);
  int*            fill       = (int*)carve(Ec * 4);
  int*            num_tiles  = (int*)carve(4);
  int*            tile_meta  = (int*)carve(MAXT * 3 * 4);
  int*            row_token  = (int*)carve((size_t)TKc * 4);
  float*          pos_weight = (float*)carve((size_t)TKc * 4);
  int*            pos_of     = (int*)carve((size_t)TKc * 4);
  unsigned short* h_buf      = (unsigned short*)carve((size_t)TKc * Ic * 2);
  unsigned short* shared_h   = (unsigned short*)carve((size_t)Tc * SIc * 2);
  unsigned short* down_buf   = (unsigned short*)carve((size_t)TKc * Hc * 2);

  // router + dispatch
  router_kernel<<<Tc / 8, 256, 0, stream>>>(x, router_w, router_bias, topk_idx,
                                            topk_w);
  init_kernel<<<1, 64, 0, stream>>>(counts);
  count_kernel<<<TKc / 256, 256, 0, stream>>>(topk_idx, counts);
  scan_build_kernel<<<1, 1, 0, stream>>>(counts, offsets, fill, tile_meta,
                                         num_tiles);
  scatter_kernel<<<TKc / 256, 256, 0, stream>>>(topk_idx, topk_w, fill,
                                                row_token, pos_weight, pos_of);

  // routed experts: gate+up (SwiGLU) then down (weighted)
  gateup_kernel<true><<<dim3(MAXT, Ic / 64), 256, 0, stream>>>(
      x, w_gate, w_up, tile_meta, num_tiles, row_token, h_buf, Hc, Ic, 0);
  down_kernel<true><<<dim3(MAXT, Hc / 64), 256, 0, stream>>>(
      h_buf, w_down, tile_meta, num_tiles, pos_weight, down_buf, nullptr, Ic,
      Hc, 0);

  // shared experts: gate+up then down (writes out)
  gateup_kernel<false><<<dim3(Tc / 64, SIc / 64), 256, 0, stream>>>(
      x, sw_gate, sw_up, nullptr, nullptr, nullptr, shared_h, Hc, SIc, Tc);
  down_kernel<false><<<dim3(Tc / 64, Hc / 64), 256, 0, stream>>>(
      shared_h, sw_down, nullptr, nullptr, nullptr, nullptr, out, SIc, Hc, Tc);

  // combine routed contributions into out
  combine_kernel<<<(Tc * Hc) / 256, 256, 0, stream>>>(pos_of, down_buf, out);
}
